// VanillaRNNLanguageModel_65137474011753
// MI455X (gfx1250) — matrix operand order repaired and checked by reading, not compiled, not run on hardware
//
#include <hip/hip_runtime.h>
#include <math.h>

// ---------------------------------------------------------------------------
// Types for CDNA5 WMMA (wave32, gfx1250)
// ---------------------------------------------------------------------------
typedef __bf16 bf16;
typedef __attribute__((ext_vector_type(16))) __bf16        v16bf;
typedef __attribute__((ext_vector_type(8)))  float         v8f;
typedef __attribute__((ext_vector_type(4)))  unsigned int  u32x4;
typedef __attribute__((ext_vector_type(4)))  int           i32x4;
typedef __attribute__((ext_vector_type(8)))  int           i32x8;

union ABf { v16bf v; u32x4 q[2]; };   // 32B fragment: two 16B chunks

static __device__ __forceinline__ v8f wmma_bf16(const ABf& a, const ABf& b, v8f c) {
  return __builtin_amdgcn_wmma_f32_16x16x32_bf16(
      /*neg_a=*/false, a.v, /*neg_b=*/false, b.v,
      /*c_mod=*/(short)0, c, /*reuse_a=*/false, /*reuse_b=*/false);
}

// Problem constants
#define VSZ 32000
#define HSZ 512
#define TSZ 256
#define BSZ 16

// ---------------------------------------------------------------------------
// Elementwise prep kernels
// ---------------------------------------------------------------------------
__global__ void k_f32_to_bf16(const float* __restrict__ s, bf16* __restrict__ d, long n) {
  long i = (long)blockIdx.x * blockDim.x + threadIdx.x;
  long stride = (long)gridDim.x * blockDim.x;
  for (; i < n; i += stride) d[i] = (bf16)s[i];
}

__global__ void k_bias_sum(const float* __restrict__ a, const float* __restrict__ b,
                           float* __restrict__ o, int n) {
  int i = blockIdx.x * blockDim.x + threadIdx.x;
  if (i < n) o[i] = a[i] + b[i];
}

// Embedding gather: e[(t*B+b)][:] = bf16(emb[x[b][t]][:])
__global__ void k_embed(const int* __restrict__ x, const float* __restrict__ emb,
                        bf16* __restrict__ e) {
  const int t = blockIdx.x, b = blockIdx.y;
  const int tok = x[b * TSZ + t];
  const float* src = emb + (long)tok * HSZ;
  bf16* dst = e + (long)(t * BSZ + b) * HSZ;
  for (int i = threadIdx.x; i < HSZ; i += blockDim.x) dst[i] = (bf16)src[i];
}

// ---------------------------------------------------------------------------
// Generic bf16 GEMM:  C[m][n] = sum_k A[m][k] * Bsrc[n][k]  (+ biasN[n] + biasM[m])
//   A:    row-major [Mtot, 512], lda = 512
//   Bsrc: "row n" at  Bsrc + zoff*z + n*bstride   (length-512 contiguous k)
//   C:    row-major, ldc = Ntot, base C + z*czstride
// WG: 256 thr / 8 waves (4x2), wave tile 32x32, WG tile 128x64.
// K staged by 64 through DOUBLE-BUFFERED LDS with register staging, so the
// next tile's global loads overlap the current tile's 8 WMMAs; one barrier
// per K-step.
// ---------------------------------------------------------------------------
__global__ __launch_bounds__(256) void k_gemm_bf16(
    const bf16* __restrict__ A, const bf16* __restrict__ Bsrc,
    const float* __restrict__ biasN, const float* __restrict__ biasM,
    float* __restrict__ C, int Ntot, long bstride, long zoff, long czstride) {
  __shared__ bf16 sA[2][128 * 72];   // 128 rows x 64 k (+8 pad)   x2 buffers
  __shared__ bf16 sB[2][64 * 72];    // 64  rows x 64 k (+8 pad)   x2 buffers

  const int tid = threadIdx.x;
  const int wv = tid >> 5, l = tid & 31, ln = l & 15, q = l >> 4;
  const int wm = wv & 3, wn = wv >> 2;
  const long Mbase = (long)blockIdx.x * 128;
  const int  Nbase = blockIdx.y * 64;
  const bf16*  Bz = Bsrc + zoff * blockIdx.z;
  float*       Cz = C + czstride * blockIdx.z;

  // Per-thread staging coordinates (A: 4 chunks of 8 bf16, B: 2 chunks)
  int aRow[4], aK[4], bRow[2], bK[2];
  const bf16* aPtr[4];
  const bf16* bPtr[2];
#pragma unroll
  for (int i = 0; i < 4; ++i) {
    int c = tid + i * 256;
    aRow[i] = c >> 3; aK[i] = (c & 7) * 8;
    aPtr[i] = &A[(Mbase + aRow[i]) * 512 + aK[i]];
  }
#pragma unroll
  for (int i = 0; i < 2; ++i) {
    int c = tid + i * 256;
    bRow[i] = c >> 3; bK[i] = (c & 7) * 8;
    bPtr[i] = &Bz[(long)(Nbase + bRow[i]) * bstride + bK[i]];
  }

  v8f acc[2][2] = {};

  // Prologue: stage K-tile 0 into buffer 0
  u32x4 ra[4], rb[2];
#pragma unroll
  for (int i = 0; i < 4; ++i) ra[i] = *(const u32x4*)(aPtr[i]);
#pragma unroll
  for (int i = 0; i < 2; ++i) rb[i] = *(const u32x4*)(bPtr[i]);
#pragma unroll
  for (int i = 0; i < 4; ++i) *(u32x4*)&sA[0][aRow[i] * 72 + aK[i]] = ra[i];
#pragma unroll
  for (int i = 0; i < 2; ++i) *(u32x4*)&sB[0][bRow[i] * 72 + bK[i]] = rb[i];

#pragma unroll
  for (int ki = 0; ki < 8; ++ki) {
    const int cur = ki & 1;
    // Issue next tile's global loads (in flight during compute)
    if (ki < 7) {
      const int kb = (ki + 1) * 64;
#pragma unroll
      for (int i = 0; i < 4; ++i) ra[i] = *(const u32x4*)(aPtr[i] + kb);
#pragma unroll
      for (int i = 0; i < 2; ++i) rb[i] = *(const u32x4*)(bPtr[i] + kb);
    }
    __syncthreads();   // buffer `cur` now fully populated; prev reads done

    const bf16* sAc = sA[cur];
    const bf16* sBc = sB[cur];
#pragma unroll
    for (int ks = 0; ks < 64; ks += 32) {
      ABf af[2], bfg[2];
#pragma unroll
      for (int s = 0; s < 2; ++s) {
        // A frag (16x32 bf16): lane ln = row m; K chunk [q*16, q*16+16)
        const bf16* ap = &sAc[(wm * 32 + s * 16 + ln) * 72 + ks];
        af[s].q[0] = *(const u32x4*)(ap + q * 16);
        af[s].q[1] = *(const u32x4*)(ap + q * 16 + 8);
        // B frag (32x16 bf16): lane ln = col n; K chunk [q*16, q*16+16)
        const bf16* bp = &sBc[(wn * 32 + s * 16 + ln) * 72 + ks + q * 16];
        bfg[s].q[0] = *(const u32x4*)(bp);
        bfg[s].q[1] = *(const u32x4*)(bp + 8);
      }
#pragma unroll
      for (int sm = 0; sm < 2; ++sm)
#pragma unroll
        for (int sn = 0; sn < 2; ++sn)
          acc[sm][sn] = wmma_bf16(af[sm], bfg[sn], acc[sm][sn]);
    }

    // Store staged registers into the other buffer (visible after next barrier)
    if (ki < 7) {
      const int nxt = cur ^ 1;
#pragma unroll
      for (int i = 0; i < 4; ++i) *(u32x4*)&sA[nxt][aRow[i] * 72 + aK[i]] = ra[i];
#pragma unroll
      for (int i = 0; i < 2; ++i) *(u32x4*)&sB[nxt][bRow[i] * 72 + bK[i]] = rb[i];
    }
  }

  // Epilogue: C layout = VGPR r, lane l -> M = r + 8*(l/16), N = l%16
#pragma unroll
  for (int sm = 0; sm < 2; ++sm) {
#pragma unroll
    for (int sn = 0; sn < 2; ++sn) {
      const int n = Nbase + wn * 32 + sn * 16 + ln;
      const float bn = biasN ? biasN[n] : 0.0f;
#pragma unroll
      for (int r = 0; r < 8; ++r) {
        const long m = Mbase + wm * 32 + sm * 16 + r + 8 * q;
        const float bm = biasM ? biasM[m] : 0.0f;
        Cz[m * (long)Ntot + n] = acc[sm][sn][r] + bn + bm;
      }
    }
  }
}

// ---------------------------------------------------------------------------
// RNN recurrence (one layer): h_t = tanh(xw_t + h_{t-1} @ w_hh^T), t = 0..255
// Single workgroup: 1024 thr / 32 waves; wave w owns H-columns [16w, 16w+16).
// w_hh bf16 split: K 0..255 resident in LDS (staged by the Tensor Data Mover,
// with D# padding reproducing the 528B conflict-free row stride), K 256..511
// in per-wave register fragments. h double-buffered in LDS; one barrier/step.
// ---------------------------------------------------------------------------
__global__ __launch_bounds__(1024) void k_rnn_recur(
    const float* __restrict__ xw,    // [T*B][512], biases pre-added
    const bf16* __restrict__ whh,    // [512][512] bf16
    bf16* __restrict__ hout) {       // [T*B][512] bf16 (A-ready for next GEMM)
  __shared__ bf16 sBlo[512 * 264];   // [col][k 0..255] (+16B pad) = 270,336 B
  __shared__ bf16 sH[2][16 * 520];   // [buf][b][k] (+8 pad)       =  33,280 B

  const int tid = threadIdx.x;
  const int wv = tid >> 5, l = tid & 31, ln = l & 15, q = l >> 4;
  const int col = wv * 16 + ln;

  // --- TDM: stage w_hh K-low half (512 rows x 512B) into padded LDS. ---
  // D# (ISA 8.3-8.6): 2D tensor, data_size=4B, tile 128x512 elements,
  // tensor_dim0_stride=256 (=1KB row), pad 16B after every 512B stored.
  if (tid < 32) {   // wave 0 issues the DMA (EXEC ignored by TDM)
    const unsigned ldsoff = (unsigned)(size_t)(&sBlo[0]);
    const unsigned long long ga = (unsigned long long)(size_t)whh;
    u32x4 g0;
    g0[0] = 1u;                                        // count=1, user mode
    g0[1] = ldsoff;                                    // lds_addr
    g0[2] = (unsigned)(ga & 0xffffffffu);              // global_addr[31:0]
    g0[3] = (unsigned)((ga >> 32) & 0x01ffffffu) | (2u << 30);  // [56:32]|type=2
    i32x8 g1;
    g1[0] = (int)((2u << 16)      // data_size = 4B
                | (1u << 20)      // pad_enable
                | (6u << 22)      // pad_interval: 128 dwords (512B)
                | (3u << 25));    // pad_amount:   4 dwords (16B)
    g1[1] = (int)(256u << 16);    // tensor_dim0[15:0]=256 (1KB row, 4B units)
    g1[2] = (int)(512u << 16);    // tensor_dim1[15:0]=512 rows
    g1[3] = (int)(128u << 16);    // tile_dim0=128 (512B of each row)
    g1[4] = (int)512;             // tile_dim1=512, tile_dim2=0
    g1[5] = (int)256;             // tensor_dim0_stride[31:0]=256
    g1[6] = 0; g1[7] = 0;         // tensor_dim1_stride=0 (2D)
    i32x4 gz = {0, 0, 0, 0};
#if defined(__clang_major__) && (__clang_major__ >= 23)
    i32x8 gz8 = {0, 0, 0, 0, 0, 0, 0, 0};
    __builtin_amdgcn_tensor_load_to_lds(g0, g1, gz, gz, gz8, 0);
#else
    __builtin_amdgcn_tensor_load_to_lds(g0, g1, gz, gz, 0);
#endif
    __builtin_amdgcn_s_wait_tensorcnt(0);
  }

  // K-high half as per-wave register B fragments (8 x 32B = 64 VGPRs)
  ABf bhi[8];
#pragma unroll
  for (int kt = 0; kt < 8; ++kt) {
    const bf16* s = whh + (long)col * 512 + 256 + kt * 32 + q * 16;
    bhi[kt].q[0] = *(const u32x4*)(s);
    bhi[kt].q[1] = *(const u32x4*)(s + 8);
  }
  // h_{-1} = 0
  for (int i = tid; i < 16 * 520; i += 1024) sH[0][i] = (bf16)0.0f;
  __syncthreads();   // TDM done (wave0 waited) + zeros visible

  for (int t = 0; t < TSZ; ++t) {
    const int cur = t & 1, nxt = cur ^ 1;
    // acc init from precomputed input projection (C layout)
    v8f acc0, acc1 = {};
#pragma unroll
    for (int r = 0; r < 8; ++r)
      acc0[r] = xw[(long)(t * BSZ + r + 8 * q) * 512 + col];
    // Prefetch next step's xw rows while this step computes
    if (t + 1 < TSZ)
      __builtin_prefetch(&xw[(long)((t + 1) * BSZ + 8 * q) * 512 + col], 0, 3);

    // K 0..255 from LDS
#pragma unroll
    for (int kt = 0; kt < 8; ++kt) {
      ABf a, b;
      const bf16* hr = &sH[cur][ln * 520 + kt * 32];
      a.q[0] = *(const u32x4*)(hr + q * 16);
      a.q[1] = *(const u32x4*)(hr + q * 16 + 8);
      const bf16* br = &sBlo[col * 264 + kt * 32 + q * 16];
      b.q[0] = *(const u32x4*)(br);
      b.q[1] = *(const u32x4*)(br + 8);
      if (kt & 1) acc1 = wmma_bf16(a, b, acc1);
      else        acc0 = wmma_bf16(a, b, acc0);
    }
    // K 256..511 from registers
#pragma unroll
    for (int kt = 8; kt < 16; ++kt) {
      ABf a;
      const bf16* hr = &sH[cur][ln * 520 + kt * 32];
      a.q[0] = *(const u32x4*)(hr + q * 16);
      a.q[1] = *(const u32x4*)(hr + q * 16 + 8);
      if (kt & 1) acc1 = wmma_bf16(a, bhi[kt - 8], acc1);
      else        acc0 = wmma_bf16(a, bhi[kt - 8], acc0);
    }

    // tanh, publish to double buffer + global
#pragma unroll
    for (int r = 0; r < 8; ++r) {
      const int m = r + 8 * q;                 // batch row 0..15
      const bf16 hv = (bf16)tanhf(acc0[r] + acc1[r]);
      sH[nxt][m * 520 + col] = hv;
      hout[(long)(t * BSZ + m) * 512 + col] = hv;
    }
    __syncthreads();
  }
}

// ---------------------------------------------------------------------------
// Host-side orchestration
// ---------------------------------------------------------------------------
extern "C" void kernel_launch(void* const* d_in, const int* in_sizes, int n_in,
                              void* d_out, int out_size, void* d_ws, size_t ws_size,
                              hipStream_t stream) {
  const int*   x     = (const int*)d_in[0];     // [B,T]
  const float* emb   = (const float*)d_in[1];   // [V,E]
  const float* w_ih  = (const float*)d_in[2];   // [L,H,H]
  const float* w_hh  = (const float*)d_in[3];   // [L,H,H]
  const float* b_ih  = (const float*)d_in[4];   // [L,H]
  const float* b_hh  = (const float*)d_in[5];   // [L,H]
  const float* w_out = (const float*)d_in[6];   // [V,H]
  const float* b_out = (const float*)d_in[7];   // [V]
  float* out = (float*)d_out;                   // [B,V,T]

  // Workspace carve-up (all 256B aligned)
  char* ws = (char*)d_ws;
  size_t off = 0;
  auto carve = [&](size_t bytes) { char* p = ws + off; off = (off + bytes + 255) & ~(size_t)255; return p; };
  bf16*  wihb  = (bf16*)carve(2 * 512 * 512 * sizeof(bf16));
  bf16*  whhb  = (bf16*)carve(2 * 512 * 512 * sizeof(bf16));
  bf16*  woutb = (bf16*)carve((size_t)VSZ * 512 * sizeof(bf16));
  float* bias01 = (float*)carve(2 * 512 * sizeof(float));
  bf16*  eb    = (bf16*)carve((size_t)TSZ * BSZ * 512 * sizeof(bf16));
  float* xw    = (float*)carve((size_t)TSZ * BSZ * 512 * sizeof(float));
  bf16*  h0b   = (bf16*)carve((size_t)TSZ * BSZ * 512 * sizeof(bf16));
  bf16*  h1b   = (bf16*)carve((size_t)TSZ * BSZ * 512 * sizeof(bf16));
  (void)ws_size; (void)n_in; (void)in_sizes; (void)out_size;

  // 1) Precision prep: weights -> bf16, combined RNN bias
  k_f32_to_bf16<<<1024, 256, 0, stream>>>(w_ih, wihb, 2L * 512 * 512);
  k_f32_to_bf16<<<1024, 256, 0, stream>>>(w_hh, whhb, 2L * 512 * 512);
  k_f32_to_bf16<<<4096, 256, 0, stream>>>(w_out, woutb, (long)VSZ * 512);
  k_bias_sum<<<4, 256, 0, stream>>>(b_ih, b_hh, bias01, 2 * 512);

  // 2) Embedding gather (padding_idx handled by emb row 0 == 0)
  k_embed<<<dim3(TSZ, BSZ), 256, 0, stream>>>(x, emb, eb);

  // 3) Layer 0: input projection (M=4096,N=512) then recurrence
  k_gemm_bf16<<<dim3(32, 8, 1), 256, 0, stream>>>(
      eb, wihb, bias01, nullptr, xw, 512, /*bstride=*/512, /*zoff=*/0, /*czstride=*/0);
  k_rnn_recur<<<1, 1024, 0, stream>>>(xw, whhb, h0b);

  // 4) Layer 1
  k_gemm_bf16<<<dim3(32, 8, 1), 256, 0, stream>>>(
      h0b, wihb + 512 * 512, bias01 + 512, nullptr, xw, 512, 512, 0, 0);
  k_rnn_recur<<<1, 1024, 0, stream>>>(xw, whhb + 512 * 512, h1b);

  // 5) LM head: per batch b, C_b[V,T] = Wout * h_b^T + b_out  -> out[b][v][t]
  //    B "row t" = h1b + b*512 + t*(16*512); stores are T-contiguous.
  k_gemm_bf16<<<dim3(VSZ / 128, TSZ / 64, BSZ), 256, 0, stream>>>(
      woutb, h1b, /*biasN=*/nullptr, /*biasM=*/b_out, out,
      /*Ntot=*/TSZ, /*bstride=*/(long)BSZ * 512, /*zoff=*/512,
      /*czstride=*/(long)VSZ * TSZ);
}
